// MultiModalMamba_48120813584732
// MI455X (gfx1250) — matrix operand order repaired and checked by reading, not compiled, not run on hardware
//
#include <hip/hip_runtime.h>
#include <hip/hip_bf16.h>

// ---------------------------------------------------------------------------
// Problem constants (from the reference)
// ---------------------------------------------------------------------------
constexpr int kB      = 8;
constexpr int kTimg   = 64;
constexpr int kTq     = 4096;
constexpr int kQdim   = 242;
constexpr int kD      = 512;
constexpr int kNH     = 8;
constexpr int kHD     = 64;
constexpr int kMaxKV  = 512;
constexpr int kV      = 1296;          // NUM_CLASSES
constexpr int kFin    = 3 * 112 * 112; // 37632

// WMMA vector types (gfx1250, wave32)
typedef __attribute__((ext_vector_type(16))) __bf16 v16bf;
typedef __attribute__((ext_vector_type(4)))  __bf16 v4bf;
typedef __attribute__((ext_vector_type(8)))  float  v8f;
typedef __attribute__((ext_vector_type(4)))  float  v4f;
typedef __attribute__((__vector_size__(16))) int    v4i;

// CDNA5 async global->LDS staging (probe via __has_builtin; fallback = VGPR path)
#if defined(__HIP_DEVICE_COMPILE__) && __has_builtin(__builtin_amdgcn_global_load_async_to_lds_b128)
#define ASYNC_LDS 1
#else
#define ASYNC_LDS 0
#endif

#define AS1 __attribute__((address_space(1)))
#define AS3 __attribute__((address_space(3)))

__device__ __forceinline__ void async_wait_all() {
#if defined(__HIP_DEVICE_COMPILE__) && __has_builtin(__builtin_amdgcn_s_wait_asynccnt)
    __builtin_amdgcn_s_wait_asynccnt(0);
#endif
}

// ---------------------------------------------------------------------------
// Generic f32-in / f32-out GEMM via bf16 WMMA.
//   C[M,N] = op(A[M,K] @ B[K,N] (+ bias)) (+ C)
// Block tile 64x128, K-step 32, 8 waves (2x4), each wave 32x32 (2x2 WMMA).
// LDS holds f32 tiles (A row-major [m][k], B row-major [k][n]), double-buffered;
// fast path stages via async global->LDS; converts to bf16 at fragment build
// with native vector fptrunc (RNE).
// flags: 1 = add bias[col], 2 = relu, 4 = accumulate into existing C
// ---------------------------------------------------------------------------
#define GEMM_BM 64
#define GEMM_BN 128
#define GEMM_BK 32

// Unpredicated (interior) tile staging: async direct-to-LDS if available.
__device__ __forceinline__ void stage_fast(
    const float* __restrict__ A, const float* __restrict__ Bm,
    float* tA, float* tB, int lda, int ldb,
    int blockRow, int blockCol, int k0, int tid, bool ldaVec)
{
#if ASYNC_LDS
    if (ldaVec) {
        #pragma unroll
        for (int e = 0; e < 2; ++e) {                 // A: 64x32 f32, b128 x2/thread
            const int t   = tid + e * 256;
            const int row = t >> 3;
            const int c   = (t & 7) * 4;
            const float* g = A + (long long)(blockRow + row) * lda + k0 + c;
            __builtin_amdgcn_global_load_async_to_lds_b128(
                (AS1 v4i*)g, (AS3 v4i*)&tA[row * GEMM_BK + c], 0, 0);
        }
    } else {
        #pragma unroll
        for (int e = 0; e < 8; ++e) {                 // rows only 4B-aligned: b32 x8
            const int t   = tid + e * 256;
            const int row = t >> 5;
            const int c   = t & 31;
            const float* g = A + (long long)(blockRow + row) * lda + k0 + c;
            __builtin_amdgcn_global_load_async_to_lds_b32(
                (AS1 int*)g, (AS3 int*)&tA[row * GEMM_BK + c], 0, 0);
        }
    }
    #pragma unroll
    for (int e = 0; e < 4; ++e) {                     // B: 32x128 f32, b128 x4/thread
        const int t  = tid + e * 256;
        const int kk = t >> 5;
        const int nn = (t & 31) * 4;
        const float* g = Bm + (long long)(k0 + kk) * ldb + blockCol + nn;
        __builtin_amdgcn_global_load_async_to_lds_b128(
            (AS1 v4i*)g, (AS3 v4i*)&tB[kk * GEMM_BN + nn], 0, 0);
    }
#else
    if (ldaVec) {
        #pragma unroll
        for (int e = 0; e < 2; ++e) {
            const int t   = tid + e * 256;
            const int row = t >> 3;
            const int c   = (t & 7) * 4;
            const float4 v = *(const float4*)(A + (long long)(blockRow + row) * lda + k0 + c);
            *(float4*)&tA[row * GEMM_BK + c] = v;
        }
    } else {
        #pragma unroll
        for (int e = 0; e < 8; ++e) {
            const int t   = tid + e * 256;
            const int row = t >> 5;
            const int c   = t & 31;
            tA[row * GEMM_BK + c] = A[(long long)(blockRow + row) * lda + k0 + c];
        }
    }
    #pragma unroll
    for (int e = 0; e < 4; ++e) {
        const int t  = tid + e * 256;
        const int kk = t >> 5;
        const int nn = (t & 31) * 4;
        const float4 v = *(const float4*)(Bm + (long long)(k0 + kk) * ldb + blockCol + nn);
        *(float4*)&tB[kk * GEMM_BN + nn] = v;
    }
#endif
}

// Guarded (edge / K-tail) staging with zero fill.
__device__ __forceinline__ void stage_guarded(
    const float* __restrict__ A, const float* __restrict__ Bm,
    float* tA, float* tB, int M, int N, int K, int lda, int ldb,
    int blockRow, int blockCol, int k0, int tid)
{
    {
        const int row = tid >> 2;
        const int col = (tid & 3) * 8;
        const int gr  = blockRow + row;
        #pragma unroll
        for (int e = 0; e < 8; ++e) {
            const int gk = k0 + col + e;
            tA[row * GEMM_BK + col + e] =
                (gr < M && gk < K) ? A[(long long)gr * lda + gk] : 0.0f;
        }
    }
    #pragma unroll
    for (int e = 0; e < 16; ++e) {
        const int linear = tid + e * 256;
        const int kk = linear >> 7;
        const int nn = linear & 127;
        const int gk = k0 + kk;
        const int gn = blockCol + nn;
        tB[kk * GEMM_BN + nn] =
            (gk < K && gn < N) ? Bm[(long long)gk * ldb + gn] : 0.0f;
    }
}

// Fragment build (f32 LDS -> bf16 via native vector fptrunc) + 2x2 WMMA.
// Layouts per ISA 7.12.2:
//   A 16x32: lanes 0-15 hold K=0..7 then 16..23; lanes 16-31 K=8..15 then 24..31
//   B 32x16: same K order as A (N = lane&15)
__device__ __forceinline__ void compute_tile(
    const float* __restrict__ tA, const float* __restrict__ tB,
    v8f acc[2][2], int wm, int wn, int half, int l16)
{
    union Frag { v16bf v; v4bf q[4]; };
    Frag afr[2], bfr[2];
    #pragma unroll
    for (int mi = 0; mi < 2; ++mi) {
        const float* src = tA + (wm * 32 + mi * 16 + l16) * GEMM_BK + half * 8;
        afr[mi].q[0] = __builtin_convertvector(*(const v4f*)(src + 0),  v4bf);
        afr[mi].q[1] = __builtin_convertvector(*(const v4f*)(src + 4),  v4bf);
        afr[mi].q[2] = __builtin_convertvector(*(const v4f*)(src + 16), v4bf);
        afr[mi].q[3] = __builtin_convertvector(*(const v4f*)(src + 20), v4bf);
    }
    #pragma unroll
    for (int ni = 0; ni < 2; ++ni) {
        const int n    = wn * 32 + ni * 16 + l16;
        const float* src = tB + (half * 8) * GEMM_BN + n;    // stride GEMM_BN per K
        #pragma unroll
        for (int q = 0; q < 4; ++q) {
            const int kq = (q < 2) ? 4 * q : 8 + 4 * q;
            const v4f t = { src[(kq + 0) * GEMM_BN], src[(kq + 1) * GEMM_BN],
                            src[(kq + 2) * GEMM_BN], src[(kq + 3) * GEMM_BN] };
            bfr[ni].q[q] = __builtin_convertvector(t, v4bf);
        }
    }
    #pragma unroll
    for (int mi = 0; mi < 2; ++mi)
        #pragma unroll
        for (int ni = 0; ni < 2; ++ni)
            acc[mi][ni] = __builtin_amdgcn_wmma_f32_16x16x32_bf16(
                false, afr[mi].v, false, bfr[ni].v,
                (short)0, acc[mi][ni], false, false);
}

__global__ __launch_bounds__(256)
void gemm_bf16_wmma(const float* __restrict__ A, const float* __restrict__ Bm,
                    const float* __restrict__ bias, float* __restrict__ C,
                    int M, int N, int K, int lda, int ldb, int ldc, int flags)
{
    __shared__ __align__(16) float sA[2][GEMM_BM * GEMM_BK];  // 2 x 8 KB
    __shared__ __align__(16) float sB[2][GEMM_BK * GEMM_BN];  // 2 x 16 KB

    const int tid  = threadIdx.x;
    const int lane = tid & 31;
    const int wave = tid >> 5;
    const int wm   = wave >> 2;   // 0..1
    const int wn   = wave & 3;    // 0..3
    const int half = lane >> 4;
    const int l16  = lane & 15;

    const int blockRow = blockIdx.y * GEMM_BM;
    const int blockCol = blockIdx.x * GEMM_BN;

    v8f acc[2][2];
    const v8f vzero = {0.f, 0.f, 0.f, 0.f, 0.f, 0.f, 0.f, 0.f};
    acc[0][0] = vzero; acc[0][1] = vzero; acc[1][0] = vzero; acc[1][1] = vzero;

    const bool interior = (blockRow + GEMM_BM <= M) && (blockCol + GEMM_BN <= N) &&
                          ((ldb & 3) == 0);
    const bool ldaVec = ((lda & 3) == 0);
    const int  kFull  = interior ? (K & ~(GEMM_BK - 1)) : 0;

    // -------- pipelined interior body: async double-buffered staging --------
    int buf = 0;
    if (kFull > 0) {
        stage_fast(A, Bm, sA[0], sB[0], lda, ldb, blockRow, blockCol, 0, tid, ldaVec);
        for (int k0 = 0; k0 < kFull; k0 += GEMM_BK) {
            async_wait_all();          // this wave's async tile writes landed
            __syncthreads();           // everyone's landed; prev compute done
            const int nk = k0 + GEMM_BK;
            if (nk < kFull)
                stage_fast(A, Bm, sA[buf ^ 1], sB[buf ^ 1], lda, ldb,
                           blockRow, blockCol, nk, tid, ldaVec);
            compute_tile(sA[buf], sB[buf], acc, wm, wn, half, l16);
            buf ^= 1;
        }
    }
    // -------- guarded remainder (K tail and/or edge blocks) --------
    for (int k0 = kFull; k0 < K; k0 += GEMM_BK) {
        __syncthreads();
        stage_guarded(A, Bm, sA[0], sB[0], M, N, K, lda, ldb,
                      blockRow, blockCol, k0, tid);
        __syncthreads();
        compute_tile(sA[0], sB[0], acc, wm, wn, half, l16);
    }

    // -------- epilogue: VGPR e -> M = e + 8*(lane>=16); N = lane&15 --------
    #pragma unroll
    for (int mi = 0; mi < 2; ++mi) {
        #pragma unroll
        for (int ni = 0; ni < 2; ++ni) {
            const int col = blockCol + wn * 32 + ni * 16 + l16;
            if (col >= N) continue;
            const float bval = (flags & 1) ? bias[col] : 0.0f;
            #pragma unroll
            for (int e = 0; e < 8; ++e) {
                const int row = blockRow + wm * 32 + mi * 16 + half * 8 + e;
                if (row >= M) continue;
                float v = acc[mi][ni][e] + bval;
                if (flags & 2) v = fmaxf(v, 0.0f);
                const long long idx = (long long)row * ldc + col;
                if (flags & 4) v += C[idx];
                C[idx] = v;
            }
        }
    }
}

// ---------------------------------------------------------------------------
// Row-wise LayerNorm (in place), one block per row.
// ---------------------------------------------------------------------------
__global__ __launch_bounds__(256)
void layernorm_rows(float* __restrict__ X, const float* __restrict__ g,
                    const float* __restrict__ bta, int Dd)
{
    __shared__ float red[256];
    float* x = X + (long long)blockIdx.x * Dd;
    const int tid = threadIdx.x;

    float s = 0.f;
    for (int i = tid; i < Dd; i += 256) s += x[i];
    red[tid] = s; __syncthreads();
    for (int off = 128; off > 0; off >>= 1) {
        if (tid < off) red[tid] += red[tid + off];
        __syncthreads();
    }
    const float mean = red[0] / (float)Dd;
    __syncthreads();

    float v = 0.f;
    for (int i = tid; i < Dd; i += 256) { float d = x[i] - mean; v += d * d; }
    red[tid] = v; __syncthreads();
    for (int off = 128; off > 0; off >>= 1) {
        if (tid < off) red[tid] += red[tid + off];
        __syncthreads();
    }
    const float rstd = rsqrtf(red[0] / (float)Dd + 1e-5f);
    __syncthreads();

    for (int i = tid; i < Dd; i += 256)
        x[i] = (x[i] - mean) * rstd * g[i] + bta[i];
}

// ---------------------------------------------------------------------------
// a += b
// ---------------------------------------------------------------------------
__global__ __launch_bounds__(256)
void add_inplace(float* __restrict__ a, const float* __restrict__ b, long long n)
{
    for (long long i = (long long)blockIdx.x * 256 + threadIdx.x; i < n;
         i += (long long)gridDim.x * 256)
        a[i] += b[i];
}

// ---------------------------------------------------------------------------
// Ragged variable-stride mean pooling: X (B, T_Q, D) -> Y (B, MAX_KV, D)
// ---------------------------------------------------------------------------
__global__ __launch_bounds__(256)
void pool_mean(const float* __restrict__ X, const int* __restrict__ lengths,
               float* __restrict__ Y)
{
    const int b = blockIdx.x / kMaxKV;
    const int j = blockIdx.x % kMaxKV;
    int L = lengths[b];
    int s = (L + kMaxKV - 1) / kMaxKV; if (s < 1) s = 1;
    const int nl = (L + s - 1) / s;

    float* y = Y + ((long long)b * kMaxKV + j) * kD;
    if (j >= nl) {
        for (int d = threadIdx.x; d < kD; d += 256) y[d] = 0.f;
        return;
    }
    const float* xb = X + (long long)b * kTq * kD;
    const float inv = 1.0f / (float)s;
    for (int d = threadIdx.x; d < kD; d += 256) {
        float acc = 0.f;
        for (int i = 0; i < s; ++i) {
            const int idx = j * s + i;
            if (idx < L) acc += xb[(long long)idx * kD + d];
        }
        y[d] = acc * inv;
    }
}

// ---------------------------------------------------------------------------
// Fused masked cross-attention, one block per (b, h).
// ---------------------------------------------------------------------------
__global__ __launch_bounds__(256)
void cross_attention(const float* __restrict__ Q, const float* __restrict__ Kt,
                     const float* __restrict__ V, const int* __restrict__ lengths,
                     float* __restrict__ O)
{
    __shared__ float qrow[kHD];
    __shared__ float probs[kMaxKV];
    __shared__ float red[256];

    const int b = blockIdx.x / kNH;
    const int h = blockIdx.x % kNH;
    int L = lengths[b];
    int s = (L + kMaxKV - 1) / kMaxKV; if (s < 1) s = 1;
    const int nl = (L + s - 1) / s;

    const int tid = threadIdx.x;

    for (int t = 0; t < kTimg; ++t) {
        if (tid < kHD)
            qrow[tid] = Q[((long long)(b * kTimg + t) * kNH + h) * kHD + tid];
        __syncthreads();

        float sc[2];
        float mymax = -1e30f;
        #pragma unroll
        for (int e = 0; e < 2; ++e) {
            const int kk = tid * 2 + e;
            const float* kv = Kt + ((long long)(b * kMaxKV + kk) * kNH + h) * kHD;
            float dot = 0.f;
            #pragma unroll 4
            for (int d = 0; d < kHD; ++d) dot += qrow[d] * kv[d];
            dot *= 0.125f;
            if (kk >= nl) dot = -1e9f;
            sc[e] = dot;
            mymax = fmaxf(mymax, dot);
        }
        red[tid] = mymax; __syncthreads();
        for (int off = 128; off > 0; off >>= 1) {
            if (tid < off) red[tid] = fmaxf(red[tid], red[tid + off]);
            __syncthreads();
        }
        const float m = red[0];
        __syncthreads();

        float mysum = 0.f;
        #pragma unroll
        for (int e = 0; e < 2; ++e) {
            const float p = __expf(sc[e] - m);
            probs[tid * 2 + e] = p;
            mysum += p;
        }
        red[tid] = mysum; __syncthreads();
        for (int off = 128; off > 0; off >>= 1) {
            if (tid < off) red[tid] += red[tid + off];
            __syncthreads();
        }
        const float inv = 1.0f / red[0];
        __syncthreads();

        const int d = tid & 63;
        const int chunk = tid >> 6;
        float part = 0.f;
        for (int kk = chunk * 128; kk < chunk * 128 + 128; ++kk)
            part += probs[kk] * V[((long long)(b * kMaxKV + kk) * kNH + h) * kHD + d];
        red[tid] = part * inv;
        __syncthreads();
        if (tid < kHD) {
            const float o = red[tid] + red[tid + 64] + red[tid + 128] + red[tid + 192];
            O[((long long)(b * kTimg + t) * kNH + h) * kHD + tid] = o;
        }
        __syncthreads();
    }
}

// ---------------------------------------------------------------------------
// Host-side launcher (graph-capture safe)
// ---------------------------------------------------------------------------
extern "C" void kernel_launch(void* const* d_in, const int* in_sizes, int n_in,
                              void* d_out, int out_size, void* d_ws, size_t ws_size,
                              hipStream_t stream)
{
    const float* images    = (const float*)d_in[0];
    const float* qgrids    = (const float*)d_in[1];
    const float* keypoints = (const float*)d_in[2];
    const int*   qlens     = (const int*)  d_in[3];
    const float* W_img     = (const float*)d_in[4];
    const float* b_img     = (const float*)d_in[5];
    const float* W_kp      = (const float*)d_in[6];
    const float* b_kp      = (const float*)d_in[7];
    const float* g_kp      = (const float*)d_in[8];
    const float* bt_kp     = (const float*)d_in[9];
    const float* W_qp      = (const float*)d_in[10];
    const float* b_qp      = (const float*)d_in[11];
    const float* g_qln     = (const float*)d_in[12];
    const float* bt_qln    = (const float*)d_in[13];
    const float* Wq        = (const float*)d_in[14];
    const float* bq        = (const float*)d_in[15];
    const float* Wk        = (const float*)d_in[16];
    const float* bk        = (const float*)d_in[17];
    const float* Wv        = (const float*)d_in[18];
    const float* bv        = (const float*)d_in[19];
    const float* Wo        = (const float*)d_in[20];
    const float* bo        = (const float*)d_in[21];
    const float* g_attn    = (const float*)d_in[22];
    const float* bt_attn   = (const float*)d_in[23];
    const float* W_ctc     = (const float*)d_in[24];
    const float* b_ctc     = (const float*)d_in[25];
    float* out = (float*)d_out;

    float* ws     = (float*)d_ws;
    float* low    = ws;                          // 512*512
    float* kpf    = low    + 262144;             // 512*512
    float* qe     = kpf    + 262144;             // 32768*512
    float* pooled = qe     + 16777216;           // 4096*512
    float* qp     = pooled + 2097152;            // 512*512
    float* kp_    = qp     + 262144;             // 4096*512
    float* vp     = kp_    + 2097152;            // 4096*512
    float* ctx    = vp     + 2097152;            // 512*512
    float* ctxo   = ctx    + 262144;             // 512*512

    const dim3 blk(256);
    auto ggrid = [](int M, int N) {
        return dim3((unsigned)((N + GEMM_BN - 1) / GEMM_BN),
                    (unsigned)((M + GEMM_BM - 1) / GEMM_BM), 1);
    };

    const int BT = kB * kTimg;      // 512 image tokens
    const int BK = kB * kMaxKV;     // 4096 pooled tokens

    gemm_bf16_wmma<<<ggrid(BT, kD), blk, 0, stream>>>(
        images, W_img, b_img, low, BT, kD, kFin, kFin, kD, kD, 1);
    gemm_bf16_wmma<<<ggrid(BT, kD), blk, 0, stream>>>(
        keypoints, W_kp, b_kp, kpf, BT, kD, kQdim, kQdim, kD, kD, 1 | 2);
    layernorm_rows<<<BT, blk, 0, stream>>>(kpf, g_kp, bt_kp, kD);
    add_inplace<<<1024, blk, 0, stream>>>(low, kpf, (long long)BT * kD);
    gemm_bf16_wmma<<<ggrid(kB * kTq, kD), blk, 0, stream>>>(
        qgrids, W_qp, b_qp, qe, kB * kTq, kD, kQdim, kQdim, kD, kD, 1);
    layernorm_rows<<<kB * kTq, blk, 0, stream>>>(qe, g_qln, bt_qln, kD);
    pool_mean<<<kB * kMaxKV, blk, 0, stream>>>(qe, qlens, pooled);
    gemm_bf16_wmma<<<ggrid(BT, kD), blk, 0, stream>>>(
        low, Wq, bq, qp, BT, kD, kD, kD, kD, kD, 1);
    gemm_bf16_wmma<<<ggrid(BK, kD), blk, 0, stream>>>(
        pooled, Wk, bk, kp_, BK, kD, kD, kD, kD, kD, 1);
    gemm_bf16_wmma<<<ggrid(BK, kD), blk, 0, stream>>>(
        pooled, Wv, bv, vp, BK, kD, kD, kD, kD, kD, 1);
    cross_attention<<<kB * kNH, blk, 0, stream>>>(qp, kp_, vp, qlens, ctx);
    gemm_bf16_wmma<<<ggrid(BT, kD), blk, 0, stream>>>(
        ctx, Wo, bo, ctxo, BT, kD, kD, kD, kD, kD, 1);
    layernorm_rows<<<BT, blk, 0, stream>>>(ctxo, g_attn, bt_attn, kD);
    gemm_bf16_wmma<<<ggrid(BT, kV), blk, 0, stream>>>(
        low, W_ctc, b_ctc, out, BT, kV, kD, kD, kV, kV, 1);
    gemm_bf16_wmma<<<ggrid(BT, kV), blk, 0, stream>>>(
        ctxo, W_ctc + (long long)kD * kV, nullptr, out, BT, kV, kD, kD, kV, kV, 4);
    (void)in_sizes; (void)n_in; (void)out_size; (void)ws_size;
}
